// BrevitasQuantConv2d_15367392985732
// MI455X (gfx1250) — compile-verified, matrix operand order checked by reading, not run on hardware
//
#include <hip/hip_runtime.h>
#include <hip/hip_bf16.h>

typedef __attribute__((ext_vector_type(16))) _Float16 v16h;
typedef __attribute__((ext_vector_type(8)))  _Float16 v8h;
typedef __attribute__((ext_vector_type(8)))  float    v8f;

#define QMAX 7.0f
#define SMIN 2e-16f

#define C_IN   256
#define C_OUT  256
#define HW     3136     // 56*56
#define W_DIM  56
#define TAPS   9
#define NPIX   64       // pixels per block (N tile); 3136 % 64 == 0
#define BSTRIDE 264     // 256 + 8 f16 pad -> conflict-free per-lane ds_load_b128

// ---------------------------------------------------------------------------
// Kernel 1: per-output-channel int4 fake-quant of w [O=256, C=256, 3, 3],
// packed directly into WMMA 16x32 f16 A-fragment lane order:
//   wA[(((mt*9 + t)*8 + kb)*32 + lane)*16 + e]
// lane/e mapping per CDNA5 ISA 7.12.2 (16-bit A 16x32):
//   lanes 0-15 : M=lane,    K in {0..7, 16..23}
//   lanes 16-31: M=lane-16, K in {8..15, 24..31}
// ---------------------------------------------------------------------------
__global__ __launch_bounds__(256)
void quant_pack_weights(const float* __restrict__ w, _Float16* __restrict__ wA) {
    const int o   = blockIdx.x;      // output channel
    const int tid = threadIdx.x;     // = input channel c
    __shared__ float red[256];

    const float* wo = w + (size_t)o * (C_IN * TAPS);
    float vals[TAPS];
    float amax = 0.0f;
#pragma unroll
    for (int t = 0; t < TAPS; ++t) {
        vals[t] = wo[tid * TAPS + t];
        amax = fmaxf(amax, fabsf(vals[t]));
    }
    red[tid] = amax;
    __syncthreads();
#pragma unroll
    for (int s = 128; s > 0; s >>= 1) {
        if (tid < s) red[tid] = fmaxf(red[tid], red[tid + s]);
        __syncthreads();
    }
    const float scale = fmaxf(red[0] * (1.0f / QMAX), SMIN);
    const float inv   = 1.0f / scale;

    // pack position for this (o, c=tid)
    const int c    = tid;
    const int kb   = c >> 5;          // K-chunk of 32
    const int k    = c & 31;          // K within chunk
    const int mt   = o >> 4;          // 16-row M tile
    const int M    = o & 15;
    const int lane = M + ((k & 8) ? 16 : 0);
    const int kk   = k & ~8;          // {0..7, 16..23}
    const int e    = (kk & 7) | ((kk & 16) >> 1);   // element 0..15 in v16h

#pragma unroll
    for (int t = 0; t < TAPS; ++t) {
        float q = rintf(vals[t] * inv);
        q = fminf(QMAX, fmaxf(-QMAX, q)) * scale;
        wA[(size_t)(((mt * TAPS + t) * 8 + kb) * 32 + lane) * 16 + e] = (_Float16)q;
    }
}

// ---------------------------------------------------------------------------
// Kernel 2: implicit-GEMM conv. One block = 64 pixels x all 256 out-channels.
// 8 wave32s; wave wv owns M tiles {2wv, 2wv+1} and all four 16-pixel N
// subtiles -> 8 persistent v8f accumulators per wave, 8 WMMAs per K-chunk.
// ---------------------------------------------------------------------------
__global__ __launch_bounds__(256)
void conv_wmma(const float* __restrict__ x, const _Float16* __restrict__ wA,
               float* __restrict__ out) {
    __shared__ _Float16 Bs[NPIX * BSTRIDE];   // [pixel][channel] f16, padded

    const int tid  = threadIdx.x;
    const int lane = tid & 31;
    const int wv   = tid >> 5;

    const int pb   = blockIdx.x * NPIX;       // global pixel base
    const int nimg = pb / HW;                 // HW % 64 == 0: no image crossing
    const int hw0  = pb % HW;

    // ---- staging coords: thread owns 1 pixel-group (4 px) x 2 adj channels
    const int pg  = tid >> 4;                 // pixel group 0..15
    const int cp  = (tid & 15) * 2;           // even channel within 32
    const int gb  = hw0 + pg * 4;             // group base pixel (mult of 4 ->
    const int gh  = gb / W_DIM;               //  whole group in one image row)
    const int gw  = gb % W_DIM;
    const float* ximg = x + (size_t)nimg * C_IN * HW;

    // ---- per-wave fragment coordinates
    const int mt0   = wv * 2;
    const int mt1   = wv * 2 + 1;
    const int nA    = lane & 15;              // N column within 16
    const int kbase = (lane < 16) ? 0 : 8;    // B-frag K base per ISA layout

    v8f acc[2][4];
#pragma unroll
    for (int m = 0; m < 2; ++m)
#pragma unroll
        for (int h = 0; h < 4; ++h) acc[m][h] = (v8f){};

    for (int t = 0; t < TAPS; ++t) {
        // ---------------- stage B tile: 256 C x 64 pix for tap t ----------
        const int dh = t / 3 - 1, dw = t % 3 - 1;
        const int hp = gh + dh;
        const bool rowok = (unsigned)hp < (unsigned)W_DIM;
        const int rbase  = hp * W_DIM + gw + dw;
        bool vmask[4];
#pragma unroll
        for (int i = 0; i < 4; ++i)
            vmask[i] = rowok && ((unsigned)(gw + dw + i) < (unsigned)W_DIM);

#pragma unroll
        for (int j = 0; j < 8; ++j) {
            const int c0 = cp + 32 * j;
            const float* xc0 = ximg + (size_t)c0 * HW;
            const float* xc1 = xc0 + HW;
#pragma unroll
            for (int i = 0; i < 4; ++i) {
                float a = vmask[i] ? xc0[rbase + i] : 0.0f;
                float b = vmask[i] ? xc1[rbase + i] : 0.0f;
                union { _Float16 h[2]; unsigned u; } pk;
                pk.h[0] = (_Float16)a;
                pk.h[1] = (_Float16)b;
                *(unsigned*)&Bs[(pg * 4 + i) * BSTRIDE + c0] = pk.u;
            }
        }
        __syncthreads();

        // ---------------- compute: 8 K-chunks of 32 ----------------------
        const _Float16* a0 = wA + (size_t)(mt0 * TAPS + t) * 4096 + lane * 16;
        const _Float16* a1 = wA + (size_t)(mt1 * TAPS + t) * 4096 + lane * 16;
#pragma unroll
        for (int kb = 0; kb < 8; ++kb) {
            v16h A0 = *(const v16h*)(a0 + kb * 512);
            v16h A1 = *(const v16h*)(a1 + kb * 512);

            union { v16h v; v8h h[2]; } bf[4];
#pragma unroll
            for (int h = 0; h < 4; ++h) {
                const _Float16* bp =
                    &Bs[(h * 16 + nA) * BSTRIDE + kb * 32 + kbase];
                bf[h].h[0] = *(const v8h*)bp;        // K = kbase..kbase+7
                bf[h].h[1] = *(const v8h*)(bp + 16); // K = kbase+16..+23
            }
#pragma unroll
            for (int h = 0; h < 4; ++h) {
                acc[0][h] = __builtin_amdgcn_wmma_f32_16x16x32_f16(
                    false, A0, false, bf[h].v, (short)0, acc[0][h], false, false);
                acc[1][h] = __builtin_amdgcn_wmma_f32_16x16x32_f16(
                    false, A1, false, bf[h].v, (short)0, acc[1][h], false, false);
            }
        }
        __syncthreads();   // protect Bs before next tap's staging
    }

    // ---------------- epilogue: scatter C/D tiles to NCHW -----------------
    // C/D layout: VGPR r -> M = r + 8*(lane>=16), N = lane%16
    const int rowOff = (lane < 16) ? 0 : 8;
    float* oimg = out + (size_t)nimg * C_OUT * HW;
#pragma unroll
    for (int m = 0; m < 2; ++m) {
        const int ob = (m == 0 ? mt0 : mt1) * 16 + rowOff;
#pragma unroll
        for (int h = 0; h < 4; ++h) {
            const int hwp = hw0 + h * 16 + (lane & 15);
#pragma unroll
            for (int r = 0; r < 8; ++r)
                oimg[(size_t)(ob + r) * HW + hwp] = acc[m][h][r];
        }
    }
}

// ---------------------------------------------------------------------------
extern "C" void kernel_launch(void* const* d_in, const int* in_sizes, int n_in,
                              void* d_out, int out_size, void* d_ws, size_t ws_size,
                              hipStream_t stream) {
    const float* x = (const float*)d_in[0];   // [32,256,56,56]
    const float* w = (const float*)d_in[1];   // [256,256,3,3]
    float* out = (float*)d_out;               // [32,256,56,56]
    _Float16* wA = (_Float16*)d_ws;           // packed quantized weights, ~1.13 MB

    quant_pack_weights<<<C_OUT, 256, 0, stream>>>(w, wA);

    const int nblocks = (32 * HW) / NPIX;     // 1568
    conv_wmma<<<nblocks, 256, 0, stream>>>(x, wA, out);
}
